// QEinsum_57415122813418
// MI455X (gfx1250) — 8-bit matrix operand order repaired and checked by reading, compiled, not run on hardware
//
#include <hip/hip_runtime.h>


typedef int v8i __attribute__((ext_vector_type(8)));

#define K_DIM 256
#define N_DIM 256
#define LDS_STRIDE 272
#define BQ_OFFSET 256

__global__ void init_ws_kernel(unsigned* ws) {
  if (threadIdx.x < 2) ws[threadIdx.x] = 0u;
}

__global__ __launch_bounds__(256)
void maxabs_kernel(const float* __restrict__ x, int n4, unsigned* __restrict__ out_bits) {
  const float4* x4 = (const float4*)x;
  float m = 0.0f;
  for (int i = blockIdx.x * blockDim.x + threadIdx.x; i < n4;
       i += gridDim.x * blockDim.x) {
    float4 v = x4[i];
    m = fmaxf(m, fmaxf(fmaxf(fabsf(v.x), fabsf(v.y)),
                       fmaxf(fabsf(v.z), fabsf(v.w))));
  }
  __shared__ float red[256];
  red[threadIdx.x] = m;
  __syncthreads();
  for (int s = 128; s > 0; s >>= 1) {
    if (threadIdx.x < s)
      red[threadIdx.x] = fmaxf(red[threadIdx.x], red[threadIdx.x + s]);
    __syncthreads();
  }
  if (threadIdx.x == 0) atomicMax(out_bits, __float_as_uint(red[0]));
}

__device__ __forceinline__ float slot_to_scale(unsigned bits) {
  return fmaxf(__uint_as_float(bits) * (1.0f / 127.0f), 1e-8f);
}

__global__ void quant_rhs_kernel(const float* __restrict__ rhs,
                                 const unsigned* __restrict__ ws_bits,
                                 signed char* __restrict__ bq) {
  float inv = 1.0f / slot_to_scale(ws_bits[1]);
  int d = blockIdx.x;
  int f = threadIdx.x;
  int q = __float2int_rn(rhs[d * N_DIM + f] * inv);
  q = q > 127 ? 127 : (q < -127 ? -127 : q);
  bq[f * K_DIM + d] = (signed char)q;
}

__device__ __forceinline__ int pack_q4(float4 v, float inv) {
  int q0 = __float2int_rn(v.x * inv);
  int q1 = __float2int_rn(v.y * inv);
  int q2 = __float2int_rn(v.z * inv);
  int q3 = __float2int_rn(v.w * inv);
  q0 = min(127, max(-127, q0));
  q1 = min(127, max(-127, q1));
  q2 = min(127, max(-127, q2));
  q3 = min(127, max(-127, q3));
  return (q0 & 255) | ((q1 & 255) << 8) | ((q2 & 255) << 16) | ((q3 & 255) << 24);
}

__global__ __launch_bounds__(256)
void gemm_kernel(const float* __restrict__ lhs, const signed char* __restrict__ bq,
                 const unsigned* __restrict__ ws_bits, float* __restrict__ out) {
  __shared__ unsigned char ldsB[N_DIM * LDS_STRIDE];

  {
    const uint4* src = (const uint4*)bq;
    for (int c = threadIdx.x; c < N_DIM * (K_DIM / 16); c += blockDim.x) {
      int n  = c >> 4;
      int ko = (c & 15) << 4;
      *(uint4*)&ldsB[n * LDS_STRIDE + ko] = src[c];
    }
  }

  float s_lhs = slot_to_scale(ws_bits[0]);
  float s_rhs = slot_to_scale(ws_bits[1]);
  float inv_a = 1.0f / s_lhs;
  float deq   = s_lhs * s_rhs;

  __syncthreads();

  const int wave = threadIdx.x >> 5;
  const int lane = threadIdx.x & 31;
  const int half = lane >> 4;
  const int l16  = lane & 15;

  const int mt = wave & 3;
  const int nh = wave >> 2;

  const long m_lane = (long)blockIdx.x * 64 + mt * 16 + l16;
  const float* arow = lhs + m_lane * K_DIM;

  v8i acc[8];
  #pragma unroll
  for (int nt = 0; nt < 8; ++nt) acc[nt] = (v8i){0, 0, 0, 0, 0, 0, 0, 0};

  #pragma unroll
  for (int kk = 0; kk < 4; ++kk) {
    v8i a;
    #pragma unroll
    for (int c = 0; c < 4; ++c) {
      int kbase = kk * 64 + (c >> 1) * 32 + (c & 1) * 8 + half * 16;
      float4 f0 = *(const float4*)(arow + kbase);
      float4 f1 = *(const float4*)(arow + kbase + 4);
      a[2 * c]     = pack_q4(f0, inv_a);
      a[2 * c + 1] = pack_q4(f1, inv_a);
    }
    #pragma unroll
    for (int nt = 0; nt < 8; ++nt) {
      int n = nh * 128 + nt * 16 + l16;
      const unsigned char* bp = &ldsB[n * LDS_STRIDE + kk * 64 + half * 16];
      uint4 lo = *(const uint4*)bp;
      uint4 hi = *(const uint4*)(bp + 32);
      v8i b;
      b[0] = lo.x; b[1] = lo.y; b[2] = lo.z; b[3] = lo.w;
      b[4] = hi.x; b[5] = hi.y; b[6] = hi.z; b[7] = hi.w;
      acc[nt] = __builtin_amdgcn_wmma_i32_16x16x64_iu8(
           true, a,  true, b, acc[nt],
           false,  false);
    }
  }

  const long mbase = (long)blockIdx.x * 64 + mt * 16 + half * 8;
  #pragma unroll
  for (int nt = 0; nt < 8; ++nt) {
    int n = nh * 128 + nt * 16 + l16;
    #pragma unroll
    for (int r = 0; r < 8; ++r) {
      out[(mbase + r) * N_DIM + n] = (float)acc[nt][r] * deq;
    }
  }
}

extern "C" void kernel_launch(void* const* d_in, const int* in_sizes, int n_in,
                              void* d_out, int out_size, void* d_ws, size_t ws_size,
                              hipStream_t stream) {
  const float* lhs = (const float*)d_in[0];
  const float* rhs = (const float*)d_in[1];
  float* out = (float*)d_out;
  unsigned* ws_bits = (unsigned*)d_ws;
  signed char* bq = (signed char*)d_ws + BQ_OFFSET;

  const int n_lhs = in_sizes[0];
  const int n_rhs = in_sizes[1];
  const int M = n_lhs / K_DIM;

  init_ws_kernel<<<1, 64, 0, stream>>>(ws_bits);
  maxabs_kernel<<<4096, 256, 0, stream>>>(lhs, n_lhs / 4, ws_bits + 0);
  maxabs_kernel<<<64, 256, 0, stream>>>(rhs, n_rhs / 4, ws_bits + 1);
  quant_rhs_kernel<<<K_DIM, N_DIM, 0, stream>>>(rhs, ws_bits, bq);
  gemm_kernel<<<M / 64, 256, 0, stream>>>(lhs, bq, ws_bits, out);
}
